// Encoder_81707457839418
// MI455X (gfx1250) — matrix operand order repaired and checked by reading, not compiled, not run on hardware
//
#include <hip/hip_runtime.h>

// ---- CDNA5 WMMA vector types ----
typedef __attribute__((ext_vector_type(16))) __bf16 v16bf;
typedef __attribute__((ext_vector_type(8)))  float  v8f;

// ---- problem constants (from reference) ----
#define VOC   32000
#define Ee    256
#define Hh    1024
#define Gg    4096      // 4*H
#define Bb    64
#define Ss    512
#define KTOT  1280      // E + H (fused [x;h] GEMM K)
#define KC_TOT 40       // KTOT / 32
#define NWG   16        // persistent workgroups; each owns 64 hidden units
#define FRAG  512       // one 16x16x32 fragment = 32 lanes * 16 bf16 elems
#define LDS_A_BYTES (4 * KC_TOT * FRAG * 2)   // 160 KB: A-frags [m 4][kc 40][1KB]

// float -> bf16, round-to-nearest-even, bit-exact control
__device__ __forceinline__ __bf16 tobf(float f) {
  unsigned u = __builtin_bit_cast(unsigned, f);
  u += 0x7FFFu + ((u >> 16) & 1u);
  unsigned short h = (unsigned short)(u >> 16);
  return __builtin_bit_cast(__bf16, h);
}

__device__ __forceinline__ float sigmoidf_(float x) {
  return 1.0f / (1.0f + __expf(-x));
}

// CDNA5 async global->LDS copy: 16 bytes per lane.
// GVS addressing: saddr = SGPR base, vaddr = per-lane u32 offset, vdst = LDS byte offset.
__device__ __forceinline__ void async_cp16(unsigned ldsoff, const void* base, unsigned goff) {
  asm volatile("global_load_async_to_lds_b128 %0, %1, %2"
               :: "v"(ldsoff), "v"(goff), "s"(base) : "memory");
}
__device__ __forceinline__ void wait_async0() {
  asm volatile("s_wait_asynccnt 0x0" ::: "memory");
}

// -------- kernel 0: zero h ping-pong fragment buffers + barrier counter --------
__global__ void k_zero(unsigned* __restrict__ p, int n) {
  int i = blockIdx.x * blockDim.x + threadIdx.x;
  if (i < n) p[i] = 0u;
}

// -------- kernel 1: pack stacked [W;U] into B-fragment order, layout [kc][nt][512] --------
// wu[(kc*256 + nt)*512 + la*16 + e] = src[k][n]
//   k = kc*32 + (e<8 ? e : e+8) + 8*(la>=16) ,  n = nt*16 + (la&15)
__global__ void k_pack_wu(const float* __restrict__ W, const float* __restrict__ U,
                          __bf16* __restrict__ wu) {
  int idx = blockIdx.x * blockDim.x + threadIdx.x;
  if (idx >= 256 * KC_TOT * FRAG) return;
  int e  = idx & 15;
  int la = (idx >> 4) & 31;
  int fc = idx >> 9;              // kc*256 + nt
  int nt = fc & 255;
  int kc = fc >> 8;
  int k  = kc * 32 + ((e < 8) ? e : e + 8) + ((la >= 16) ? 8 : 0);
  int n  = nt * 16 + (la & 15);
  float v = (k < Ee) ? W[(size_t)k * Gg + n] : U[(size_t)(k - Ee) * Gg + n];
  wu[idx] = tobf(v);
}

// -------- kernel 2: embedding gather straight into A-fragment order, per timestep --------
// xf[((s*4 + m)*8 + kc)*512 + la*16 + e] = emb[ tok[row][s] ][ K ]
//   row = m*16 + (la&15) ,  K = kc*32 + (e<8 ? e : e+8) + 8*(la>=16)
__global__ void k_pack_x(const int* __restrict__ tok, const float* __restrict__ emb,
                         __bf16* __restrict__ xf) {
  int idx = blockIdx.x * blockDim.x + threadIdx.x;
  if (idx >= Ss * 4 * 8 * FRAG) return;
  int e  = idx & 15;
  int la = (idx >> 4) & 31;
  int kc = (idx >> 9) & 7;
  int m  = (idx >> 12) & 3;
  int s  = idx >> 14;
  int row = m * 16 + (la & 15);
  int K   = kc * 32 + ((e < 8) ? e : e + 8) + ((la >= 16) ? 8 : 0);
  int t = tok[row * Ss + s];
  xf[idx] = tobf(emb[(size_t)t * Ee + K]);
}

// -------- kernel 3: persistent LSTM scan --------
// 16 WGs x 256 threads (8 waves). WG wg owns hidden cols [wg*64, wg*64+64).
// Wave w owns M-tile m = w>>1 and two adjacent 16-col subtiles ns0, ns0+1 (ns0 = (w&1)*2):
// one LDS A-fragment feeds 8 WMMAs (2 positions x 4 gates). A fragments for the whole
// step (x part from precomputed layout, h part from the global ping-pong buffer) are
// staged into 160KB LDS via async global->LDS DMA once per step.
__global__ __launch_bounds__(256) void k_lstm_scan(
    const __bf16* __restrict__ wu,   // [40 kc][256 nt][512] B-frags (L2-resident, 10MB)
    const __bf16* __restrict__ xf,   // [S][4 m][8 kc][512]  A-frags (x part)
    __bf16* hfrag0, __bf16* hfrag1,  // [4 m][32 kc][512]    A-frags (h part), ping-pong
    const float* __restrict__ bias,  // [4096]
    float* __restrict__ out,         // outputs[B,S,H] ++ h_last[B,H] ++ c_last[B,H]
    unsigned* counter)
{
  extern __shared__ char lds_raw[];              // 160KB, offset 0 in LDS
  const __bf16* lds_a = (const __bf16*)lds_raw;

  const int wg   = blockIdx.x;        // 0..15
  const int tid  = threadIdx.x;
  const int wave = tid >> 5;          // 0..7
  const int lane = tid & 31;
  const int l15  = lane & 15;
  const int hi   = (lane >= 16) ? 1 : 0;
  const int j0   = wg * 64;
  const int m    = wave >> 1;         // batch-row tile 0..3
  const int ns0  = (wave & 1) * 2;    // first of two 16-col subtiles

  // B-fragment offsets for the 8 accumulators: q = pp*4 + gi
  unsigned bo[8];
  #pragma unroll
  for (int q = 0; q < 8; ++q) {
    int pp = q >> 2, gi = q & 3;
    int nt = gi * 64 + wg * 4 + ns0 + pp;       // global 16-col tile in [0,256)
    bo[q] = (unsigned)nt * FRAG + (unsigned)lane * 16;
  }

  v8f   cst[2];
  float bv[2][4];
  #pragma unroll
  for (int pp = 0; pp < 2; ++pp) {
    #pragma unroll
    for (int e = 0; e < 8; ++e) cst[pp][e] = 0.0f;
    #pragma unroll
    for (int gi = 0; gi < 4; ++gi)
      bv[pp][gi] = bias[gi * Hh + j0 + (ns0 + pp) * 16 + l15];
  }

  const __bf16* lam = lds_a + (size_t)m * KC_TOT * FRAG + (size_t)lane * 16;

  for (int t = 0; t < Ss; ++t) {
    const __bf16* hr = (t & 1) ? hfrag1 : hfrag0;
    __bf16*       hw = (t & 1) ? hfrag0 : hfrag1;

    // ---- stage this step's A fragments into LDS (async DMA, 160KB) ----
    {
      const char* xsrc = (const char*)xf + (size_t)t * 32768;   // [m][kc<8], 8KB per m
      const char* hsrc = (const char*)hr;                       // [m][kch],  32KB per m
      #pragma unroll
      for (int mm = 0; mm < 4; ++mm) {
        unsigned dbase = (unsigned)(mm * (KC_TOT * FRAG * 2));  // mm*40960
        const char* xs = xsrc + (size_t)mm * 8192;
        #pragma unroll
        for (int it = 0; it < 2; ++it) {
          unsigned o = (unsigned)((it * 256 + tid) * 16);
          async_cp16(dbase + o, xs, o);
        }
        const char* hs = hsrc + (size_t)mm * 32768;
        #pragma unroll
        for (int it = 0; it < 8; ++it) {
          unsigned o = (unsigned)((it * 256 + tid) * 16);
          async_cp16(dbase + 8192u + o, hs, o);
        }
      }
      wait_async0();
      __syncthreads();
    }

    // ---- fused [x;h] @ [W;U] over K=1280: 1 LDS A-frag -> 8 WMMAs ----
    v8f acc[8];
    #pragma unroll
    for (int q = 0; q < 8; ++q)
      #pragma unroll
      for (int e = 0; e < 8; ++e) acc[q][e] = bv[q >> 2][q & 3];

    for (int kc = 0; kc < KC_TOT; ++kc) {
      v16bf a = *(const v16bf*)(lam + (size_t)kc * FRAG);
      const __bf16* wk = wu + (size_t)kc * 256 * FRAG;
      v16bf bq[8];
      #pragma unroll
      for (int q = 0; q < 8; ++q) bq[q] = *(const v16bf*)(wk + bo[q]);
      #pragma unroll
      for (int q = 0; q < 8; ++q)
        acc[q] = __builtin_amdgcn_wmma_f32_16x16x32_bf16(
            false, a, false, bq[q], (short)0, acc[q], false, false);
    }

    // ---- gates + state update + stores (C/D layout: M = e + 8*hi, N = l15) ----
    #pragma unroll
    for (int pp = 0; pp < 2; ++pp) {
      int ns = ns0 + pp;
      #pragma unroll
      for (int e = 0; e < 8; ++e) {
        float iv = sigmoidf_(acc[pp * 4 + 0][e]);
        float fv = sigmoidf_(acc[pp * 4 + 1][e]);
        float gv = tanhf(acc[pp * 4 + 2][e]);
        float ov = sigmoidf_(acc[pp * 4 + 3][e]);
        float cn = fv * cst[pp][e] + iv * gv;
        float hn = ov * tanhf(cn);
        cst[pp][e] = cn;

        int rowt = e + 8 * hi;
        int row  = m * 16 + rowt;
        int col  = j0 + ns * 16 + l15;
        out[((size_t)row * Ss + t) * Hh + col] = hn;

        // scatter h into next-step A-fragment layout
        int lc  = ns * 16 + l15;          // local hidden col 0..63
        int kw  = lc & 31;
        int kch = wg * 2 + (lc >> 5);     // h k-chunk 0..31
        int la  = rowt + ((kw & 8) ? 16 : 0);
        int sl  = (kw & 7) + ((kw >= 16) ? 8 : 0);
        hw[(size_t)(m * 32 + kch) * FRAG + la * 16 + sl] = tobf(hn);

        if (t == Ss - 1) {
          out[(size_t)Bb * Ss * Hh + (size_t)row * Hh + col] = hn;                    // h_last
          out[(size_t)Bb * Ss * Hh + (size_t)Bb * Hh + (size_t)row * Hh + col] = cn;  // c_last
        }
      }
    }

    // warm GL2 for next step's x fragments
    if (t + 1 < Ss)
      __builtin_prefetch((const char*)xf + (size_t)(t + 1) * 32768 + tid * 16, 0, 1);

    // ---- device-scope grid barrier ----
    __threadfence();                    // make h stores visible (all waves)
    __syncthreads();
    if (tid == 0) {
      __hip_atomic_fetch_add(counter, 1u, __ATOMIC_RELEASE, __HIP_MEMORY_SCOPE_AGENT);
      unsigned tgt = (unsigned)(NWG * (t + 1));
      while (__hip_atomic_load(counter, __ATOMIC_ACQUIRE, __HIP_MEMORY_SCOPE_AGENT) < tgt)
        __builtin_amdgcn_s_sleep(1);
    }
    __syncthreads();
    __threadfence();                    // discard stale cached h before next read
  }
}

extern "C" void kernel_launch(void* const* d_in, const int* in_sizes, int n_in,
                              void* d_out, int out_size, void* d_ws, size_t ws_size,
                              hipStream_t stream) {
  const int*   tokens = (const int*)  d_in[0];   // [B,S]
  const float* emb    = (const float*)d_in[1];   // [V,E]
  const float* W      = (const float*)d_in[2];   // [E,4H]
  const float* U      = (const float*)d_in[3];   // [H,4H]
  const float* bias   = (const float*)d_in[4];   // [4H]
  float* out = (float*)d_out;

  char* ws = (char*)d_ws;
  const size_t WU_ELEMS = (size_t)256 * KC_TOT * FRAG;   // 5,242,880 bf16 = 10.0 MiB
  const size_t XF_ELEMS = (size_t)Ss * 4 * 8 * FRAG;     // 8,388,608 bf16 = 16.0 MiB
  const size_t HF_ELEMS = (size_t)4 * 32 * FRAG;         //    65,536 bf16 = 128 KiB each
  __bf16* wu = (__bf16*)ws;
  __bf16* xf = wu + WU_ELEMS;
  __bf16* h0 = xf + XF_ELEMS;
  __bf16* h1 = h0 + HF_ELEMS;
  unsigned* counter = (unsigned*)(h1 + HF_ELEMS);

  int zN = (int)(2 * HF_ELEMS / 2) + 64;                 // hfrags (+counter) in u32s
  k_zero<<<(zN + 255) / 256, 256, 0, stream>>>((unsigned*)h0, zN);

  k_pack_wu<<<(int)(WU_ELEMS / 256), 256, 0, stream>>>(W, U, wu);
  k_pack_x <<<(int)(XF_ELEMS / 256), 256, 0, stream>>>(tokens, emb, xf);
  k_lstm_scan<<<NWG, 256, LDS_A_BYTES, stream>>>(wu, xf, h0, h1, bias, out, counter);
}
